// BaselineGCN_36429912604731
// MI455X (gfx1250) — matrix operand order repaired and checked by reading, not compiled, not run on hardware
//
#include <hip/hip_runtime.h>

// ---------------------------------------------------------------------------
// 2-layer GCN for MI455X (gfx1250).
// GEMMs run on v_wmma_f32_16x16x32_bf16 (bf16 in, fp32 accumulate).
// Graph aggregation via float atomics (L2-resident: h fits in 192MB L2).
// ---------------------------------------------------------------------------

typedef __attribute__((ext_vector_type(16))) __bf16 v16bf;
typedef __attribute__((ext_vector_type(8)))  float  v8f;

static __device__ __forceinline__ unsigned short f2bf(float x) {
  // round-to-nearest-even fp32 -> bf16
  unsigned u = __float_as_uint(x);
  unsigned r = u + 0x7fffu + ((u >> 16) & 1u);
  return (unsigned short)(r >> 16);
}

// ---- degree / norm prep ----------------------------------------------------

__global__ void k_fill1(float* __restrict__ p, int n) {
  int i = blockIdx.x * blockDim.x + threadIdx.x;
  if (i < n) p[i] = 1.0f;  // self-loop weight
}

__global__ void k_deg_accum(const int* __restrict__ dst, const float* __restrict__ ew,
                            float* __restrict__ deg, int E) {
  int e = blockIdx.x * blockDim.x + threadIdx.x;
  if (e < E) atomicAdd(&deg[dst[e]], ew[e]);
}

__global__ void k_finalize_deg(float* __restrict__ deg_then_dinv,
                               float* __restrict__ invdeg, int n) {
  int i = blockIdx.x * blockDim.x + threadIdx.x;
  if (i < n) {
    float d = deg_then_dinv[i];
    deg_then_dinv[i] = rsqrtf(d);   // in-place: becomes dinv
    invdeg[i] = 1.0f / d;
  }
}

__global__ void k_edge_norm(const int* __restrict__ src, const int* __restrict__ dst,
                            const float* __restrict__ ew, const float* __restrict__ dinv,
                            float* __restrict__ norm, int E) {
  int e = blockIdx.x * blockDim.x + threadIdx.x;
  if (e < E) norm[e] = dinv[src[e]] * ew[e] * dinv[dst[e]];
}

// ---- precision conversion --------------------------------------------------

__global__ void k_f32_to_bf16(const float* __restrict__ in, unsigned short* __restrict__ out,
                              long n, int relu) {
  long i = (long)blockIdx.x * blockDim.x + threadIdx.x;
  if (i < n) {
    float v = in[i];
    if (relu) v = fmaxf(v, 0.0f);
    out[i] = f2bf(v);
  }
}

// W: [K, F] fp32 row-major  ->  WT: [F, K] bf16 row-major (B fragments read rows of WT)
__global__ void k_transpose_bf16(const float* __restrict__ W, unsigned short* __restrict__ WT,
                                 int K, int F) {
  long i = (long)blockIdx.x * blockDim.x + threadIdx.x;
  if (i < (long)K * F) {
    int k = (int)(i / F);
    int f = (int)(i % F);
    WT[(long)f * K + k] = f2bf(W[i]);
  }
}

// ---- WMMA GEMM: C[M,F] = A[M,K] (bf16) * BT[F,K]^T (bf16), fp32 accumulate -
// One wave per 16(M) x 64(N) tile: 4 accumulators, K stepped by 32.
__global__ void __launch_bounds__(256)
k_gemm_bf16_wmma(const unsigned short* __restrict__ A,
                 const unsigned short* __restrict__ BT,
                 float* __restrict__ C, int M, int K, int F) {
  const int lane   = threadIdx.x & 31;
  const int wave   = threadIdx.x >> 5;
  const int nTiles = F >> 6;                        // 64-wide N tiles
  const int tiles  = (M >> 4) * nTiles;
  const int tile   = blockIdx.x * 8 + wave;
  if (tile >= tiles) return;                        // uniform per wave: EXEC stays full

  const int mT = tile / nTiles, nT = tile % nTiles;
  const int m0 = mT << 4, n0 = nT << 6;
  const int ml = lane & 15;                         // M (for A) or N (for B/C) within tile
  const int hi = (lane >> 4) & 1;                   // lane-half selector

  // A: lane holds row m0+ml; K = 16*hi + 0..15 contiguous (matches B).
  const unsigned* arow = (const unsigned*)(A + (size_t)(m0 + ml) * K);
  // B: lane holds column n0+16j+ml of B == row of BT; K = 16*hi + 0..15 contiguous.
  const unsigned short* b0 = BT + (size_t)(n0 +  0 + ml) * K + (hi << 4);
  const unsigned short* b1 = BT + (size_t)(n0 + 16 + ml) * K + (hi << 4);
  const unsigned short* b2 = BT + (size_t)(n0 + 32 + ml) * K + (hi << 4);
  const unsigned short* b3 = BT + (size_t)(n0 + 48 + ml) * K + (hi << 4);

  v8f acc0 = {}, acc1 = {}, acc2 = {}, acc3 = {};

  for (int k = 0; k < K; k += 32) {
    // Build A fragment (contiguous K order, same as B):
    //   vgpr j (dword): K-pair index = j + 8*hi
    union { v16bf v; unsigned u[8]; } fa;
    const unsigned* ap = arow + (k >> 1);
#pragma unroll
    for (int j = 0; j < 8; ++j)
      fa.u[j] = ap[j + (hi << 3)];

    v16bf vb0 = *(const v16bf*)(const void*)(b0 + k);
    v16bf vb1 = *(const v16bf*)(const void*)(b1 + k);
    v16bf vb2 = *(const v16bf*)(const void*)(b2 + k);
    v16bf vb3 = *(const v16bf*)(const void*)(b3 + k);

    if (k + 32 < K) {  // hint next K-chunk into cache
      __builtin_prefetch((const void*)(b0 + k + 32), 0, 0);
      __builtin_prefetch((const void*)(ap + 16), 0, 0);
    }

    acc0 = __builtin_amdgcn_wmma_f32_16x16x32_bf16(false, fa.v, false, vb0,
                                                   (short)0, acc0, false, false);
    acc1 = __builtin_amdgcn_wmma_f32_16x16x32_bf16(false, fa.v, false, vb1,
                                                   (short)0, acc1, false, false);
    acc2 = __builtin_amdgcn_wmma_f32_16x16x32_bf16(false, fa.v, false, vb2,
                                                   (short)0, acc2, false, false);
    acc3 = __builtin_amdgcn_wmma_f32_16x16x32_bf16(false, fa.v, false, vb3,
                                                   (short)0, acc3, false, false);
  }

  // C/D layout: vgpr v -> row m0 + v + 8*hi, column n0 + 16*j + ml
  const int mrow = m0 + (hi << 3);
#pragma unroll
  for (int v = 0; v < 8; ++v) {
    float* cr = C + (size_t)(mrow + v) * F + n0 + ml;
    cr[0]  = acc0[v];
    cr[16] = acc1[v];
    cr[32] = acc2[v];
    cr[48] = acc3[v];
  }
}

// ---- aggregation -----------------------------------------------------------

// out[n,f] = h[n,f] * invdeg[n]   (self-loop term, also initializes the buffer)
__global__ void k_selfloop_init(const float* __restrict__ h, const float* __restrict__ invdeg,
                                float* __restrict__ out, long total, int F) {
  long i = (long)blockIdx.x * blockDim.x + threadIdx.x;
  if (i < total) out[i] = h[i] * invdeg[i / F];
}

// out[dst[e], f..f+3] += norm[e] * h[src[e], f..f+3]
__global__ void k_scatter_add(const float* __restrict__ h, const int* __restrict__ src,
                              const int* __restrict__ dst, const float* __restrict__ norm,
                              float* __restrict__ out, int E, int F) {
  long tid = (long)blockIdx.x * blockDim.x + threadIdx.x;
  int perEdge = F >> 2;
  if (tid >= (long)E * perEdge) return;
  int e  = (int)(tid / perEdge);
  int f4 = (int)(tid % perEdge) << 2;
  float w = norm[e];
  const float4 hv = *(const float4*)(h + (size_t)src[e] * F + f4);
  float* o = out + (size_t)dst[e] * F + f4;
  atomicAdd(o + 0, w * hv.x);
  atomicAdd(o + 1, w * hv.y);
  atomicAdd(o + 2, w * hv.z);
  atomicAdd(o + 3, w * hv.w);
}

// ---------------------------------------------------------------------------

extern "C" void kernel_launch(void* const* d_in, const int* in_sizes, int n_in,
                              void* d_out, int out_size, void* d_ws, size_t ws_size,
                              hipStream_t stream) {
  const int IN = 512, HID = 512, OUT = 256;
  const float* x  = (const float*)d_in[0];   // [N, 512]
  const int*   ei = (const int*)d_in[1];     // [2, E] (harness ABI: integer -> int32)
  const float* ew = (const float*)d_in[2];   // [E]
  const float* W1 = (const float*)d_in[3];   // [512, 512]
  const float* W2 = (const float*)d_in[4];   // [512, 256]
  const int N = in_sizes[0] / IN;            // 10000
  const int E = in_sizes[2];                 // 160000
  const int* src = ei;
  const int* dst = ei + E;

  // workspace carve-out (256B aligned); total ~53 MB
  char* w = (char*)d_ws;
  auto carve = [&](size_t bytes) -> char* {
    char* p = w; w += (bytes + 255) & ~(size_t)255; return p;
  };
  float*          dinv   = (float*)carve((size_t)N * 4);           // deg -> dinv (in place)
  float*          invdeg = (float*)carve((size_t)N * 4);
  float*          norm   = (float*)carve((size_t)E * 4);
  unsigned short* xbf    = (unsigned short*)carve((size_t)N * IN * 2);  // reused for relu(h1) bf16
  unsigned short* w1t    = (unsigned short*)carve((size_t)IN * HID * 2);
  unsigned short* w2t    = (unsigned short*)carve((size_t)HID * OUT * 2);
  float*          h1     = (float*)carve((size_t)N * HID * 4);     // reused as h2
  float*          agg1   = (float*)carve((size_t)N * HID * 4);
  float*          h2     = h1;
  float*          out    = (float*)d_out;

  const int T = 256;
  auto blocks = [](long n, int t) { return (unsigned)((n + t - 1) / t); };

  // 1) degrees + edge norms
  k_fill1<<<blocks(N, T), T, 0, stream>>>(dinv, N);
  k_deg_accum<<<blocks(E, T), T, 0, stream>>>(dst, ew, dinv, E);
  k_finalize_deg<<<blocks(N, T), T, 0, stream>>>(dinv, invdeg, N);
  k_edge_norm<<<blocks(E, T), T, 0, stream>>>(src, dst, ew, dinv, norm, E);

  // 2) bf16 operands
  k_f32_to_bf16<<<blocks((long)N * IN, T), T, 0, stream>>>(x, xbf, (long)N * IN, 0);
  k_transpose_bf16<<<blocks((long)IN * HID, T), T, 0, stream>>>(W1, w1t, IN, HID);
  k_transpose_bf16<<<blocks((long)HID * OUT, T), T, 0, stream>>>(W2, w2t, HID, OUT);

  // 3) layer 1: h1 = x @ W1 (WMMA), agg1 = D^-1 h1 + scatter(norm * h1[src])
  {
    int tiles = (N >> 4) * (HID >> 6);
    k_gemm_bf16_wmma<<<blocks(tiles, 8), 256, 0, stream>>>(xbf, w1t, h1, N, IN, HID);
    k_selfloop_init<<<blocks((long)N * HID, T), T, 0, stream>>>(h1, invdeg, agg1,
                                                                (long)N * HID, HID);
    k_scatter_add<<<blocks((long)E * (HID >> 2), T), T, 0, stream>>>(h1, src, dst, norm,
                                                                     agg1, E, HID);
  }

  // 4) relu + convert for layer 2 input (reuses xbf slot)
  k_f32_to_bf16<<<blocks((long)N * HID, T), T, 0, stream>>>(agg1, xbf, (long)N * HID, 1);

  // 5) layer 2: h2 = relu(h1agg) @ W2 (WMMA), out = D^-1 h2 + scatter(norm * h2[src])
  {
    int tiles = (N >> 4) * (OUT >> 6);
    k_gemm_bf16_wmma<<<blocks(tiles, 8), 256, 0, stream>>>(xbf, w2t, h2, N, HID, OUT);
    k_selfloop_init<<<blocks((long)N * OUT, T), T, 0, stream>>>(h2, invdeg, out,
                                                                (long)N * OUT, OUT);
    k_scatter_add<<<blocks((long)E * (OUT >> 2), T), T, 0, stream>>>(h2, src, dst, norm,
                                                                     out, E, OUT);
  }
}
